// OIMLoss_36532991820638
// MI455X (gfx1250) — matrix operand order repaired and checked by reading, not compiled, not run on hardware
//
#include <hip/hip_runtime.h>

#define P_N 5532
#define Q_N 5000
#define D_N 256
#define B_N 8192
#define NB_N (P_N + Q_N)          // 10532
#define SCALAR_V 30.0f

#define MTILE 64                   // rows of X per workgroup
#define NCHUNK 32                  // bank rows per iteration
#define BSTRIDE 264                // padded LDS row stride (elements) -> conflict-free b128

typedef __attribute__((ext_vector_type(16))) __bf16 v16bf;
typedef __attribute__((ext_vector_type(8)))  __bf16 v8bf;
typedef __attribute__((ext_vector_type(4)))  __bf16 v4bf;
typedef __attribute__((ext_vector_type(8)))  float  v8f;

__device__ __forceinline__ float bftrunc_f(float x) {
    unsigned u = __builtin_bit_cast(unsigned, x) & 0xffff0000u;
    return __builtin_bit_cast(float, u);
}
__device__ __forceinline__ __bf16 bfraw(float x) {
    unsigned u = __builtin_bit_cast(unsigned, x);
    return __builtin_bit_cast(__bf16, (unsigned short)(u >> 16));
}

// ---------------------------------------------------------------- init / scan
__global__ void oim_init(int* tbl) {
    int i = blockIdx.x * 256 + threadIdx.x;
    if (i < NB_N) tbl[i] = -1;
}

__global__ void oim_scan(const int* __restrict__ label, int* __restrict__ tbl) {
    int i = blockIdx.x * 256 + threadIdx.x;
    if (i < B_N) {
        int y = label[i];
        if (y >= 0 && y < NB_N) atomicMax(&tbl[y], i);
    }
}

// --------------------------------------------- fused GEMM + online softmax lse
__global__ __launch_bounds__(128) void oim_gemm_softmax(
    const float* __restrict__ inputs, const float* __restrict__ lut,
    const float* __restrict__ cq, float* __restrict__ lse)
{
    __shared__ __bf16 Bh[NCHUNK * BSTRIDE];
    __shared__ __bf16 Bl[NCHUNK * BSTRIDE];

    const int tid  = threadIdx.x;
    const int wave = tid >> 5;
    const int lane = tid & 31;
    const int lr   = lane & 15;
    const int half = lane >> 4;

    const int m0 = blockIdx.x * MTILE + wave * 16;

    // ---- A tile (16 rows x 256) as bf16 hi/lo fragments, kept in registers.
    // 16-bit A layout: lanes 0-15 -> M=lane, K = 16h .. 16h+15 (contiguous), h=lane>>4
    v16bf a_hi[8], a_lo[8];
    {
        const float* row = inputs + (size_t)(m0 + lr) * D_N;
        #pragma unroll
        for (int c = 0; c < 8; ++c) {
            const int k0 = c * 32 + half * 16;
            alignas(16) float f[16];
            *(float4*)&f[0]  = *(const float4*)&row[k0];
            *(float4*)&f[4]  = *(const float4*)&row[k0 + 4];
            *(float4*)&f[8]  = *(const float4*)&row[k0 + 8];
            *(float4*)&f[12] = *(const float4*)&row[k0 + 12];
            #pragma unroll
            for (int e = 0; e < 16; ++e) {
                float h = bftrunc_f(f[e]);
                a_hi[c][e] = bfraw(h);
                a_lo[c][e] = bfraw(f[e] - h);
            }
        }
    }

    // per-lane online softmax state (merged across lanes once at the end)
    float m_run[8], s_run[8];
    #pragma unroll
    for (int r = 0; r < 8; ++r) { m_run[r] = -3.0e38f; s_run[r] = 0.0f; }

    const int r_ld  = tid >> 2;        // 0..31 : bank row this thread stages
    const int c0_ld = (tid & 3) * 64;  // 64 consecutive elements

    // ---- software pipeline: stage next chunk in registers during compute
    float4 stage[16];
    {   // chunk 0 (rows 0..31 always < P)
        const float* src = lut + (size_t)r_ld * D_N;
        #pragma unroll
        for (int q = 0; q < 16; ++q) stage[q] = *(const float4*)&src[c0_ld + q * 4];
    }

    const int NITER = (NB_N + NCHUNK - 1) / NCHUNK;   // 330
    for (int it = 0; it < NITER; ++it) {
        const int n0 = it * NCHUNK;
        __syncthreads();
        // ---- convert staged f32 -> bf16 hi/lo into LDS
        {
            __bf16* dh = &Bh[r_ld * BSTRIDE];
            __bf16* dl = &Bl[r_ld * BSTRIDE];
            #pragma unroll
            for (int q = 0; q < 16; ++q) {
                float4 v = stage[q];
                float h0 = bftrunc_f(v.x), h1 = bftrunc_f(v.y);
                float h2 = bftrunc_f(v.z), h3 = bftrunc_f(v.w);
                v4bf vh = { bfraw(h0), bfraw(h1), bfraw(h2), bfraw(h3) };
                v4bf vl = { bfraw(v.x - h0), bfraw(v.y - h1),
                            bfraw(v.z - h2), bfraw(v.w - h3) };
                *(v4bf*)&dh[c0_ld + q * 4] = vh;
                *(v4bf*)&dl[c0_ld + q * 4] = vl;
            }
        }
        __syncthreads();

        // ---- issue next chunk's global loads (consumed next iteration)
        if (it + 1 < NITER) {
            const int jr = n0 + NCHUNK + r_ld;
            if (jr < NB_N) {
                const float* src = (jr < P_N) ? (lut + (size_t)jr * D_N)
                                              : (cq + (size_t)(jr - P_N) * D_N);
                #pragma unroll
                for (int q = 0; q < 16; ++q) stage[q] = *(const float4*)&src[c0_ld + q * 4];
            } else {
                #pragma unroll
                for (int q = 0; q < 16; ++q) stage[q] = make_float4(0.f, 0.f, 0.f, 0.f);
            }
        }

        // ---- 2 N-tiles of 16, K=256 in 8 chunks, 3 WMMAs per (chunk, tile)
        v8f acc0 = {0,0,0,0,0,0,0,0};
        v8f acc1 = {0,0,0,0,0,0,0,0};
        #pragma unroll
        for (int c = 0; c < 8; ++c) {
            // 16-bit B layout: lane holds N=lane&15, K = 16h .. 16h+15 (contiguous)
            const int kb = c * 32 + half * 16;
            #pragma unroll
            for (int nt = 0; nt < 2; ++nt) {
                const int ro = (nt * 16 + lr) * BSTRIDE + kb;
                v8bf h0 = *(const v8bf*)&Bh[ro];
                v8bf h1 = *(const v8bf*)&Bh[ro + 8];
                v8bf l0 = *(const v8bf*)&Bl[ro];
                v8bf l1 = *(const v8bf*)&Bl[ro + 8];
                v16bf bh = __builtin_shufflevector(h0, h1, 0,1,2,3,4,5,6,7,8,9,10,11,12,13,14,15);
                v16bf bl = __builtin_shufflevector(l0, l1, 0,1,2,3,4,5,6,7,8,9,10,11,12,13,14,15);
                v8f& acc = nt ? acc1 : acc0;
                acc = __builtin_amdgcn_wmma_f32_16x16x32_bf16(false, a_hi[c], false, bh,
                                                              (short)0, acc, false, false);
                acc = __builtin_amdgcn_wmma_f32_16x16x32_bf16(false, a_hi[c], false, bl,
                                                              (short)0, acc, false, false);
                acc = __builtin_amdgcn_wmma_f32_16x16x32_bf16(false, a_lo[c], false, bh,
                                                              (short)0, acc, false, false);
            }
        }

        // ---- per-lane online softmax update (no cross-lane traffic here)
        const bool v0ok = (n0 + lr) < NB_N;
        const bool v1ok = (n0 + 16 + lr) < NB_N;
        #pragma unroll
        for (int r = 0; r < 8; ++r) {
            float v0 = v0ok ? SCALAR_V * acc0[r] : -3.0e38f;
            float v1 = v1ok ? SCALAR_V * acc1[r] : -3.0e38f;
            float mn = fmaxf(m_run[r], fmaxf(v0, v1));
            float e  = __expf(v0 - mn) + __expf(v1 - mn);
            s_run[r] = s_run[r] * __expf(m_run[r] - mn) + e;
            m_run[r] = mn;
        }
    }

    // ---- merge lanes within each 16-lane group (rows r / r+8 per half)
    #pragma unroll
    for (int r = 0; r < 8; ++r) {
        float m = m_run[r], s = s_run[r];
        #pragma unroll
        for (int st = 1; st < 16; st <<= 1) {
            float mo = __shfl_xor(m, st, 32);
            float so = __shfl_xor(s, st, 32);
            float mn = fmaxf(m, mo);
            s = s * __expf(m - mn) + so * __expf(mo - mn);
            m = mn;
        }
        if (lr == 0) lse[m0 + r + 8 * half] = m + __logf(s);
    }
}

// ---------------------------------------------------------- label logit -> nll
__global__ __launch_bounds__(256) void oim_label_nll(
    const float* __restrict__ inputs, const int* __restrict__ label,
    const float* __restrict__ lut, const float* __restrict__ cq,
    const float* __restrict__ lse, float* __restrict__ nll)
{
    const int wave = threadIdx.x >> 5, lane = threadIdx.x & 31;
    const int i = blockIdx.x * 8 + wave;
    const int y = label[i];
    const bool valid = (y >= 0) && (y < NB_N);
    const float* brow = (y >= 0 && y < P_N) ? (lut + (size_t)y * D_N)
                      : (valid ? (cq + (size_t)(y - P_N) * D_N) : lut);
    const float* xrow = inputs + (size_t)i * D_N;
    const int k0 = lane * 8;
    float4 b0 = *(const float4*)&brow[k0], b1 = *(const float4*)&brow[k0 + 4];
    float4 x0 = *(const float4*)&xrow[k0], x1 = *(const float4*)&xrow[k0 + 4];
    float d = b0.x*x0.x + b0.y*x0.y + b0.z*x0.z + b0.w*x0.w
            + b1.x*x1.x + b1.y*x1.y + b1.z*x1.z + b1.w*x1.w;
    #pragma unroll
    for (int s = 1; s < 32; s <<= 1) d += __shfl_xor(d, s, 32);
    if (lane == 0) nll[i] = valid ? (lse[i] - SCALAR_V * d) : 0.0f;
}

// -------------------------------------- deterministic fixed-order loss reduce
__global__ __launch_bounds__(256) void oim_loss_reduce(
    const float* __restrict__ nll, const int* __restrict__ label, float* __restrict__ out)
{
    __shared__ float sm[256];
    __shared__ float sc[256];
    const int tid = threadIdx.x;
    float s = 0.0f, c = 0.0f;
    for (int i = tid; i < B_N; i += 256) {
        s += nll[i];
        c += (label[i] >= 0) ? 1.0f : 0.0f;
    }
    sm[tid] = s; sc[tid] = c;
    __syncthreads();
    for (int st = 128; st > 0; st >>= 1) {
        if (tid < st) { sm[tid] += sm[tid + st]; sc[tid] += sc[tid + st]; }
        __syncthreads();
    }
    if (tid == 0) out[0] = sm[0] / fmaxf(sc[0], 1.0f);
}

// -------------------------------------------- momentum bank update (per row)
__global__ __launch_bounds__(256) void oim_update(
    const float* __restrict__ inputs, const float* __restrict__ lut,
    const float* __restrict__ cq, const int* __restrict__ tbl,
    float* __restrict__ out_lut, float* __restrict__ out_cq)
{
    const int wave = threadIdx.x >> 5, lane = threadIdx.x & 31;
    const int j = blockIdx.x * 8 + wave;
    if (j >= NB_N) return;
    const float* src = (j < P_N) ? (lut + (size_t)j * D_N)
                                 : (cq + (size_t)(j - P_N) * D_N);
    float* dst = (j < P_N) ? (out_lut + (size_t)j * D_N)
                           : (out_cq + (size_t)(j - P_N) * D_N);
    const int bi = tbl[j];
    const int k0 = lane * 8;
    float v[8];
    *(float4*)&v[0] = *(const float4*)&src[k0];
    *(float4*)&v[4] = *(const float4*)&src[k0 + 4];
    if (bi >= 0) {
        const float* x = inputs + (size_t)bi * D_N;
        float4 x0 = *(const float4*)&x[k0], x1 = *(const float4*)&x[k0 + 4];
        v[0] = 0.5f*v[0] + 0.5f*x0.x;  v[1] = 0.5f*v[1] + 0.5f*x0.y;
        v[2] = 0.5f*v[2] + 0.5f*x0.z;  v[3] = 0.5f*v[3] + 0.5f*x0.w;
        v[4] = 0.5f*v[4] + 0.5f*x1.x;  v[5] = 0.5f*v[5] + 0.5f*x1.y;
        v[6] = 0.5f*v[6] + 0.5f*x1.z;  v[7] = 0.5f*v[7] + 0.5f*x1.w;
        float ss = 0.0f;
        #pragma unroll
        for (int e = 0; e < 8; ++e) ss += v[e] * v[e];
        #pragma unroll
        for (int s = 1; s < 32; s <<= 1) ss += __shfl_xor(ss, s, 32);
        const float inv = 1.0f / sqrtf(ss);
        #pragma unroll
        for (int e = 0; e < 8; ++e) v[e] *= inv;
    }
    #pragma unroll
    for (int e = 0; e < 8; ++e) dst[k0 + e] = v[e];   // d_out+1 is only 4B aligned
}

// ------------------------------------------------------------------- launcher
extern "C" void kernel_launch(void* const* d_in, const int* in_sizes, int n_in,
                              void* d_out, int out_size, void* d_ws, size_t ws_size,
                              hipStream_t stream)
{
    const float* inputs = (const float*)d_in[0];
    const int*   label  = (const int*)d_in[1];
    const float* lut    = (const float*)d_in[2];
    const float* cq     = (const float*)d_in[3];

    float* out     = (float*)d_out;
    float* out_lut = out + 1;
    float* out_cq  = out_lut + (size_t)P_N * D_N;

    float* lse = (float*)d_ws;          // B floats
    float* nll = lse + B_N;             // B floats
    int*   tbl = (int*)(nll + B_N);     // NB ints

    oim_init        <<<(NB_N + 255) / 256, 256, 0, stream>>>(tbl);
    oim_scan        <<<(B_N + 255) / 256, 256, 0, stream>>>(label, tbl);
    oim_gemm_softmax<<<B_N / MTILE, 128, 0, stream>>>(inputs, lut, cq, lse);
    oim_label_nll   <<<B_N / 8, 256, 0, stream>>>(inputs, label, lut, cq, lse, nll);
    oim_loss_reduce <<<1, 256, 0, stream>>>(nll, label, out);
    oim_update      <<<(NB_N + 7) / 8, 256, 0, stream>>>(inputs, lut, cq, tbl, out_lut, out_cq);
}
